// Attention_4045859193206
// MI455X (gfx1250) — matrix operand order repaired and checked by reading, not compiled, not run on hardware
//
#include <hip/hip_runtime.h>

typedef __attribute__((ext_vector_type(16))) _Float16 v16h;
typedef __attribute__((ext_vector_type(8)))  _Float16 v8h;
typedef __attribute__((ext_vector_type(8)))  float    v8f;
typedef __attribute__((ext_vector_type(4)))  float    v4f;
typedef __attribute__((ext_vector_type(4)))  unsigned u32x4;
typedef __attribute__((ext_vector_type(8)))  int      i32x8;
typedef __attribute__((ext_vector_type(4)))  int      i32x4;

#define B_   128
#define N_   196
#define DIM_ 512
#define H_   16
#define HD_  32
#define NQ_  208   // padded query rows (13*16)
#define NK_  224   // padded key rows (7*32)

union H16 { v16h v; v8h h[2]; };

__device__ __forceinline__ v8f wmma16(v16h a, v16h b, v8f c) {
  return __builtin_amdgcn_wmma_f32_16x16x32_f16(false, a, false, b, (short)0, c,
                                                false, false);
}

// --- CDNA5 async global->LDS copy (ASYNCcnt) ------------------------------
__device__ __forceinline__ unsigned lds_addr32(const void* p) {
  return (unsigned)(unsigned long long)p;  // flat->LDS uses addr[31:0]
}
__device__ __forceinline__ void async_b128(unsigned lds, unsigned goff,
                                           const void* base) {
  asm volatile("global_load_async_to_lds_b128 %0, %1, %2"
               :: "v"(lds), "v"(goff), "s"(base)
               : "memory");
}
__device__ __forceinline__ void wait_async0() {
  asm volatile("s_wait_asynccnt 0" ::: "memory");
}

// --- CDNA5 Tensor Data Mover: 2D f32 tile (tile0 x tile1) of a row-major
// tensor (dim0 x dim1, stride0), global -> LDS (TENSORcnt) -----------------
__device__ __forceinline__ void tdm_load_2d_f32(unsigned lds_addr,
                                                const void* gaddr,
                                                unsigned dim0, unsigned dim1,
                                                unsigned stride0,
                                                unsigned tile0, unsigned tile1) {
  unsigned long long ga = (unsigned long long)gaddr;
  u32x4 g0;
  g0[0] = 1u;                                         // count=1, user D#
  g0[1] = lds_addr;                                   // lds_addr [63:32]
  g0[2] = (unsigned)(ga & 0xffffffffu);               // global_addr [95:64]
  g0[3] = (unsigned)((ga >> 32) & 0x01ffffffu)        // global_addr [120:96]
          | (2u << 30);                               // type=2 ("image")
  i32x8 g1;
  g1[0] = (int)(2u << 16);                            // data_size=2 -> 4 bytes
  g1[1] = (int)((dim0 & 0xffffu) << 16);              // tensor_dim0 [79:48]
  g1[2] = (int)((dim0 >> 16) | ((dim1 & 0xffffu) << 16));  // dim0 hi, dim1 lo
  g1[3] = (int)((dim1 >> 16) | (tile0 << 16));        // dim1 hi, tile_dim0
  g1[4] = (int)tile1;                                 // tile_dim1, tile_dim2=0
  g1[5] = (int)stride0;                               // dim0_stride [207:160]
  g1[6] = 0;                                          // stride0 hi, stride1 lo
  g1[7] = 0;
  i32x4 z4 = {0, 0, 0, 0};
  i32x8 z8 = {0, 0, 0, 0, 0, 0, 0, 0};
  // 6-arg form (clang-23 / therock-10.0 toolchain)
  __builtin_amdgcn_tensor_load_to_lds(g0, g1, z4, z4, z8, 0);
}

// B-fragment per-lane payload: 8 f32 at p, 8 f32 at p+16 -> 16 halves
__device__ __forceinline__ v16h cvt16c(const float* p) {
  v4f f0 = *(const v4f*)(p);
  v4f f1 = *(const v4f*)(p + 4);
  v4f f2 = *(const v4f*)(p + 16);
  v4f f3 = *(const v4f*)(p + 20);
  v16h r;
#pragma unroll
  for (int e = 0; e < 4; ++e) {
    r[e]      = (_Float16)f0[e];
    r[4 + e]  = (_Float16)f1[e];
    r[8 + e]  = (_Float16)f2[e];
    r[12 + e] = (_Float16)f3[e];
  }
  return r;
}

// A-fragment per-lane payload: 8 f32 at p, 8 f32 at p+16 -> 16 halves
__device__ __forceinline__ v16h cvtA(const float* p) {
  v4f f0 = *(const v4f*)(p);
  v4f f1 = *(const v4f*)(p + 4);
  v4f f2 = *(const v4f*)(p + 16);
  v4f f3 = *(const v4f*)(p + 20);
  v16h r;
#pragma unroll
  for (int e = 0; e < 4; ++e) {
    r[e]      = (_Float16)f0[e];
    r[4 + e]  = (_Float16)f1[e];
    r[8 + e]  = (_Float16)f2[e];
    r[12 + e] = (_Float16)f3[e];
  }
  return r;
}

__device__ __forceinline__ float rmax16(float v) {
#pragma unroll
  for (int o = 1; o < 16; o <<= 1) v = fmaxf(v, __shfl_xor(v, o, 32));
  return v;
}
__device__ __forceinline__ float rsum16(float v) {
#pragma unroll
  for (int o = 1; o < 16; o <<= 1) v += __shfl_xor(v, o, 32);
  return v;
}

// ---------------------------------------------------------------------------
// Kernel 1: densify RPE bias -> bias[h][i][j] f32
// ---------------------------------------------------------------------------
__global__ __launch_bounds__(256) void bias_kernel(
    const int* __restrict__ rpe_index, const float* __restrict__ rpe_table,
    float* __restrict__ bias) {
  int t = blockIdx.x * 256 + threadIdx.x;
  if (t >= H_ * N_ * N_) return;
  int h = t / (N_ * N_);
  int p = t - h * (N_ * N_);
  bias[t] = rpe_table[rpe_index[p] * H_ + h];
}

// ---------------------------------------------------------------------------
// Kernel 2: QKV GEMM.  M = B*N = 25088, K = 512, Nout = 1536.
// Double-buffered pipeline: A tile (128x32 f32) via async-to-LDS copies,
// B tile (64x32 f32) via one TDM descriptor issued by wave 0.
// ---------------------------------------------------------------------------
__global__ __launch_bounds__(256) void qkv_gemm(
    const float* __restrict__ x, const float* __restrict__ w,
    const float* __restrict__ qkvb, _Float16* __restrict__ qh,
    _Float16* __restrict__ kh, _Float16* __restrict__ vh) {
  __shared__ float ldsX[2][128 * 32];   // 2 x 16KB
  __shared__ float ldsW[2][64 * 32];    // 2 x 8KB

  const int t = threadIdx.x;
  const int lane = t & 31, wave = t >> 5;
  const int l16 = lane & 15, hi = lane >> 4;
  const int wm = wave & 3, wn = wave >> 2;
  const int MbBlk = blockIdx.y * 128;
  const int NbBlk = blockIdx.x * 64;
  const int Mb = MbBlk + wm * 32;
  const int Nb = NbBlk + wn * 32;
  const unsigned ldsXb = lds_addr32(ldsX);
  const unsigned ldsWb = lds_addr32(ldsW);

  auto issue = [&](int kb, int p) {
#pragma unroll
    for (int r = 0; r < 4; ++r) {  // A tile: 16KB, rows of 128B
      unsigned q = (unsigned)t * 16u + (unsigned)r * 4096u;
      unsigned row = q >> 7, col = q & 127u;
      unsigned goff = ((unsigned)(MbBlk + row) * DIM_ + kb) * 4u + col;
      async_b128(ldsXb + (unsigned)p * 16384u + q, goff, x);
    }
    if (wave == 0)  // B tile: one 64x32 f32 TDM descriptor
      tdm_load_2d_f32(ldsWb + (unsigned)p * 8192u,
                      w + (size_t)NbBlk * DIM_ + kb,
                      /*dim0=*/DIM_, /*dim1=*/3 * DIM_, /*stride0=*/DIM_,
                      /*tile0=*/32, /*tile1=*/64);
  };

  v8f acc[2][2] = {};
  issue(0, 0);
  for (int it = 0; it < 16; ++it) {
    const int p = it & 1;
    wait_async0();
    if (wave == 0) __builtin_amdgcn_s_wait_tensorcnt(0);
    __syncthreads();                       // tiles for `it` visible to all
    if (it < 15) issue((it + 1) * 32, p ^ 1);  // overlap copy with compute

    v16h a[2], bf[2];
#pragma unroll
    for (int i = 0; i < 2; ++i)
      a[i] = cvtA(ldsX[p] + (wm * 32 + i * 16 + l16) * 32 + hi * 8);
#pragma unroll
    for (int j = 0; j < 2; ++j)
      bf[j] = cvt16c(ldsW[p] + (wn * 32 + j * 16 + l16) * 32 + hi * 8);
#pragma unroll
    for (int i = 0; i < 2; ++i)
#pragma unroll
      for (int j = 0; j < 2; ++j) acc[i][j] = wmma16(a[i], bf[j], acc[i][j]);
  }

  const float scale = 0.17677669529663687f;  // HD^-0.5
#pragma unroll
  for (int i = 0; i < 2; ++i)
#pragma unroll
    for (int j = 0; j < 2; ++j) {
      const int c = Nb + j * 16 + l16;
      const float bb = qkvb[c];
      const int sect = c >> 9, cc = c & 511, h = cc >> 5, d = cc & 31;
#pragma unroll
      for (int vv = 0; vv < 8; ++vv) {
        const int m = Mb + i * 16 + hi * 8 + vv;
        const int bi = m / N_, ni = m - bi * N_;
        const int bh = bi * H_ + h;
        float val = acc[i][j][vv] + bb;
        if (sect == 0)
          qh[((size_t)bh * NQ_ + ni) * HD_ + d] = (_Float16)(val * scale);
        else if (sect == 1)
          kh[((size_t)bh * NK_ + ni) * HD_ + d] = (_Float16)val;
        else
          vh[((size_t)bh * HD_ + d) * NK_ + ni] = (_Float16)val;
      }
    }
}

// ---------------------------------------------------------------------------
// Kernel 3: flash attention.  One wave = one 16-row query tile of one (b,h).
// ---------------------------------------------------------------------------
__global__ __launch_bounds__(256) void attn_kernel(
    const _Float16* __restrict__ qh, const _Float16* __restrict__ kh,
    const _Float16* __restrict__ vh, const float* __restrict__ bias,
    const int* __restrict__ mask, _Float16* __restrict__ oh) {
  __shared__ _Float16 plds[8 * 16 * 32];

  const int lane = threadIdx.x & 31, wave = threadIdx.x >> 5;
  const int l16 = lane & 15, hi = lane >> 4;

  const int gw = blockIdx.x * 8 + wave;    // 26624 wave-tiles total
  const int qt = gw % 13;
  const int bh = gw / 13;
  const int h = bh & (H_ - 1);
  const int b = bh >> 4;
  const int irow0 = qt * 16 + hi * 8;

  H16 qa;
  {
    const _Float16* qp = qh + ((size_t)bh * NQ_ + qt * 16 + l16) * HD_;
    qa.h[0] = *(const v8h*)(qp + hi * 8);
    qa.h[1] = *(const v8h*)(qp + 16 + hi * 8);
  }

  float mrow[8], lrow[8];
#pragma unroll
  for (int vv = 0; vv < 8; ++vv) { mrow[vv] = -1e30f; lrow[vv] = 0.0f; }
  v8f o0 = {}, o1 = {};
  _Float16* pl = plds + wave * (16 * 32);

  for (int kb = 0; kb < NK_; kb += 32) {
    H16 kf0, kf1;
    {
      const _Float16* kp = kh + ((size_t)bh * NK_ + kb + l16) * HD_ + hi * 8;
      kf0.h[0] = *(const v8h*)kp;
      kf0.h[1] = *(const v8h*)(kp + 16);
      kp = kh + ((size_t)bh * NK_ + kb + 16 + l16) * HD_ + hi * 8;
      kf1.h[0] = *(const v8h*)kp;
      kf1.h[1] = *(const v8h*)(kp + 16);
    }
    v8f zc = {};
    v8f s0 = wmma16(qa.v, kf0.v, zc);
    v8f s1 = wmma16(qa.v, kf1.v, zc);

    const int jk0 = kb + l16, jk1 = kb + 16 + l16;
    const bool ok0 = (jk0 < N_) && (mask[b * N_ + jk0] != 0);
    const bool ok1 = (jk1 < N_) && (mask[b * N_ + jk1] != 0);
#pragma unroll
    for (int vv = 0; vv < 8; ++vv) {
      const int i = irow0 + vv;
      const float b0 =
          (i < N_ && jk0 < N_) ? bias[((size_t)h * N_ + i) * N_ + jk0] : 0.0f;
      const float b1 =
          (i < N_ && jk1 < N_) ? bias[((size_t)h * N_ + i) * N_ + jk1] : 0.0f;
      s0[vv] = ok0 ? (s0[vv] + b0) : -1e30f;
      s1[vv] = ok1 ? (s1[vv] + b1) : -1e30f;
    }

#pragma unroll
    for (int vv = 0; vv < 8; ++vv) {
      float mx = rmax16(fmaxf(s0[vv], s1[vv]));
      float mnew = fmaxf(mrow[vv], mx);
      float corr = __expf(mrow[vv] - mnew);
      float p0 = __expf(s0[vv] - mnew);
      float p1 = __expf(s1[vv] - mnew);
      float rs = rsum16(p0 + p1);
      mrow[vv] = mnew;
      lrow[vv] = lrow[vv] * corr + rs;
      o0[vv] *= corr;
      o1[vv] *= corr;
      const int r = hi * 8 + vv;
      pl[r * 32 + l16] = (_Float16)p0;
      pl[r * 32 + 16 + l16] = (_Float16)p1;
    }
    asm volatile("s_wait_dscnt 0" ::: "memory");

    H16 pa;
    pa.h[0] = *(const v8h*)(pl + l16 * 32 + hi * 8);
    pa.h[1] = *(const v8h*)(pl + l16 * 32 + 16 + hi * 8);

    H16 vf0, vf1;
    {
      const _Float16* vp = vh + ((size_t)bh * HD_ + l16) * NK_ + kb + hi * 8;
      vf0.h[0] = *(const v8h*)vp;
      vf0.h[1] = *(const v8h*)(vp + 16);
      vp = vh + ((size_t)bh * HD_ + 16 + l16) * NK_ + kb + hi * 8;
      vf1.h[0] = *(const v8h*)vp;
      vf1.h[1] = *(const v8h*)(vp + 16);
    }
    o0 = wmma16(pa.v, vf0.v, o0);
    o1 = wmma16(pa.v, vf1.v, o1);
  }

#pragma unroll
  for (int vv = 0; vv < 8; ++vv) {
    const int i = qt * 16 + hi * 8 + vv;
    if (i < N_) {
      const float inv = 1.0f / lrow[vv];
      const size_t base = ((size_t)b * N_ + i) * DIM_ + h * HD_;
      oh[base + l16] = (_Float16)(o0[vv] * inv);
      oh[base + 16 + l16] = (_Float16)(o1[vv] * inv);
    }
  }
}

// ---------------------------------------------------------------------------
// Kernel 4: projection GEMM (double-buffered async pipeline).
// ---------------------------------------------------------------------------
__global__ __launch_bounds__(256) void proj_gemm(
    const _Float16* __restrict__ oh, const float* __restrict__ w,
    const float* __restrict__ pb, float* __restrict__ out) {
  __shared__ _Float16 ldsO[2][128 * 32];  // 2 x 8KB
  __shared__ float ldsW[2][64 * 32];      // 2 x 8KB

  const int t = threadIdx.x;
  const int lane = t & 31, wave = t >> 5;
  const int l16 = lane & 15, hi = lane >> 4;
  const int wm = wave & 3, wn = wave >> 2;
  const int MbBlk = blockIdx.y * 128;
  const int NbBlk = blockIdx.x * 64;
  const int Mb = MbBlk + wm * 32;
  const int Nb = NbBlk + wn * 32;
  const unsigned ldsOb = lds_addr32(ldsO);
  const unsigned ldsWb = lds_addr32(ldsW);

  auto issue = [&](int kb, int p) {
#pragma unroll
    for (int r = 0; r < 2; ++r) {  // A tile (f16): 8KB, rows of 64B
      unsigned q = (unsigned)t * 16u + (unsigned)r * 4096u;
      unsigned row = q >> 6, col = q & 63u;
      unsigned goff = ((unsigned)(MbBlk + row) * DIM_ + kb) * 2u + col;
      async_b128(ldsOb + (unsigned)p * 8192u + q, goff, oh);
    }
#pragma unroll
    for (int r = 0; r < 2; ++r) {  // B tile (f32): 8KB, rows of 128B
      unsigned q = (unsigned)t * 16u + (unsigned)r * 4096u;
      unsigned row = q >> 7, col = q & 127u;
      unsigned goff = ((unsigned)(NbBlk + row) * DIM_ + kb) * 4u + col;
      async_b128(ldsWb + (unsigned)p * 8192u + q, goff, w);
    }
  };

  v8f acc[2][2] = {};
  issue(0, 0);
  for (int it = 0; it < 16; ++it) {
    const int p = it & 1;
    wait_async0();
    __syncthreads();
    if (it < 15) issue((it + 1) * 32, p ^ 1);

    H16 a[2];
    v16h bf[2];
#pragma unroll
    for (int i = 0; i < 2; ++i) {
      const _Float16* pp = ldsO[p] + (wm * 32 + i * 16 + l16) * 32 + hi * 8;
      a[i].h[0] = *(const v8h*)pp;
      a[i].h[1] = *(const v8h*)(pp + 16);
    }
#pragma unroll
    for (int j = 0; j < 2; ++j)
      bf[j] = cvt16c(ldsW[p] + (wn * 32 + j * 16 + l16) * 32 + hi * 8);
#pragma unroll
    for (int i = 0; i < 2; ++i)
#pragma unroll
      for (int j = 0; j < 2; ++j) acc[i][j] = wmma16(a[i].v, bf[j], acc[i][j]);
  }

#pragma unroll
  for (int i = 0; i < 2; ++i)
#pragma unroll
    for (int j = 0; j < 2; ++j) {
      const int c = Nb + j * 16 + l16;
      const float bb = pb[c];
#pragma unroll
      for (int vv = 0; vv < 8; ++vv) {
        const int m = Mb + i * 16 + hi * 8 + vv;
        out[(size_t)m * DIM_ + c] = acc[i][j][vv] + bb;
      }
    }
}

// ---------------------------------------------------------------------------
extern "C" void kernel_launch(void* const* d_in, const int* in_sizes, int n_in,
                              void* d_out, int out_size, void* d_ws,
                              size_t ws_size, hipStream_t stream) {
  (void)in_sizes; (void)n_in; (void)out_size; (void)ws_size;

  const float* x         = (const float*)d_in[0];
  const int*   rpe_index = (const int*)d_in[1];
  const int*   mask      = (const int*)d_in[2];
  const float* qkv_w     = (const float*)d_in[3];
  const float* qkv_b     = (const float*)d_in[4];
  const float* proj_w    = (const float*)d_in[5];
  const float* proj_b    = (const float*)d_in[6];
  const float* rpe_table = (const float*)d_in[7];
  float* out = (float*)d_out;

  char* ws = (char*)d_ws;
  size_t off = 0;
  _Float16* qh = (_Float16*)(ws + off); off += (size_t)B_ * H_ * NQ_ * HD_ * 2;
  _Float16* kh = (_Float16*)(ws + off); off += (size_t)B_ * H_ * NK_ * HD_ * 2;
  _Float16* vh = (_Float16*)(ws + off); off += (size_t)B_ * H_ * HD_ * NK_ * 2;
  const size_t qkv_bytes = off;
  _Float16* oh = (_Float16*)(ws + off); off += (size_t)B_ * N_ * DIM_ * 2;
  float* bias = (float*)(ws + off);

  (void)hipMemsetAsync(ws, 0, qkv_bytes, stream);

  bias_kernel<<<(H_ * N_ * N_ + 255) / 256, 256, 0, stream>>>(rpe_index,
                                                              rpe_table, bias);
  qkv_gemm<<<dim3(24, 196), 256, 0, stream>>>(x, qkv_w, qkv_b, qh, kh, vh);
  attn_kernel<<<(B_ * H_ * 13) / 8, 256, 0, stream>>>(qh, kh, vh, bias, mask,
                                                      oh);
  proj_gemm<<<dim3(8, 196), 256, 0, stream>>>(oh, proj_w, proj_b, out);
}
